// exVDPMLP_71459665871458
// MI455X (gfx1250) — matrix operand order repaired and checked by reading, not compiled, not run on hardware
//
#include <hip/hip_runtime.h>

#define D_IN 784
#define H 256
#define BATCH 1024
#define XSTRIDE 804    // x-tile LDS row stride (dwords): 16B-aligned rows, 16-bank spread
#define MUSTRIDE 260   // mu1-tile LDS row stride (dwords)

typedef __attribute__((ext_vector_type(16))) __bf16 v16bf;
typedef __attribute__((ext_vector_type(8)))  float  v8f;

__device__ __forceinline__ float softplusf(float v) {
    return (v > 20.f) ? v : log1pf(expf(v));
}

__device__ __forceinline__ float wred(float v) {
#pragma unroll
    for (int m = 16; m >= 1; m >>= 1) v += __shfl_xor(v, m, 32);
    return v;
}

__global__ __launch_bounds__(512)
void exvdp_fused_kernel(const float* __restrict__ x,
                        const float* __restrict__ w1_mu,
                        const float* __restrict__ w1_sigma,
                        const float* __restrict__ b1_mu,
                        const float* __restrict__ b1_sigma,
                        const float* __restrict__ w2_mu,
                        const float* __restrict__ w2_sigma,
                        const float* __restrict__ b2_mu,
                        const float* __restrict__ b2_sigma,
                        float* __restrict__ out)
{
    __shared__ float sx [16 * XSTRIDE];   // x tile [16 x 784] (+ zero pad to 804)
    __shared__ float smu[16 * MUSTRIDE];  // mu1 tile [16 x 256]

    const int tid   = threadIdx.x;
    const int wave  = tid >> 5;        // 0..15 : N-tile over H; later: batch row
    const int lane  = tid & 31;
    const int lhalf = lane >> 4;
    const int l15   = lane & 15;

    const int mrow = blockIdx.x * 16;
    const int h    = (wave << 4) + l15;    // H column owned by this lane (B/C frags)

    // ======== Stage x tile to LDS once per block (cooperative, vectorized) ==========
    // zero the pad region [784, 804) of each row so Phase A needs no bounds checks
#pragma unroll 1
    for (int i = tid; i < 16 * 20; i += 512) {
        int r = i / 20, c = 784 + (i % 20);
        sx[r * XSTRIDE + c] = 0.f;
    }
    {
        const float4* xg = (const float4*)(x + (size_t)mrow * D_IN); // 16*784 = 3136 float4
#pragma unroll 1
        for (int i = tid; i < 3136; i += 512) {
            int r = i / 196, c4 = i % 196;
            float4 v = xg[i];
            *(float4*)&sx[r * XSTRIDE + c4 * 4] = v;
        }
    }
    __syncthreads();

    // ======== Phase A: mu1 tile [16 x 256] via bf16 WMMA, hi/lo split ================
    v8f acc = {};
    const float* srow = sx + l15 * XSTRIDE;  // A-fragment row for this lane (LDS)

    for (int kc = 0; kc < 25; ++kc) {        // 784 = 24*32 + 16 (pad region supplies 0s)
        const int K0 = kc * 32;
        v16bf ahi, alo, bhi, blo;
        // A fragment 16x32 bf16: element 2*j2(+1) holds K = kbase + (j2&3)*2 (+1)
#pragma unroll
        for (int j2 = 0; j2 < 8; ++j2) {
            int k  = ((j2 & 4) ? 16 : 0) + lhalf * 8 + (j2 & 3) * 2;
            float2 xv = *(const float2*)(srow + K0 + k);   // max idx 799 < 804, pad = 0
            __bf16 h0 = (__bf16)xv.x;
            __bf16 h1 = (__bf16)xv.y;
            ahi[2 * j2]     = h0;
            ahi[2 * j2 + 1] = h1;
            alo[2 * j2]     = (__bf16)(xv.x - (float)h0);
            alo[2 * j2 + 1] = (__bf16)(xv.y - (float)h1);
        }
        // prefetch next K-chunk of w1 (global_prefetch_b8)
        {
            int kp = K0 + 32 + lhalf * 16;
            kp = (kp < D_IN) ? kp : (D_IN - 1);
            __builtin_prefetch(w1_mu + (size_t)kp * H + h, 0, 1);
        }
        // B fragment 32x16 bf16: lane = N col, element j holds K = (j&8?16:0) + lhalf*8 + (j&7)
        // tail handled branch-free: clamped address + select-zero
#pragma unroll
        for (int j = 0; j < 16; ++j) {
            int kk = K0 + ((j & 8) ? 16 : 0) + lhalf * 8 + (j & 7);
            int kc2 = (kk < D_IN) ? kk : (D_IN - 1);
            float wv = w1_mu[kc2 * H + h];
            wv = (kk < D_IN) ? wv : 0.f;
            __bf16 wh = (__bf16)wv;
            bhi[j] = wh;
            blo[j] = (__bf16)(wv - (float)wh);
        }
        // near-f32 product: xhi*whi + xhi*wlo + xlo*whi, f32 accumulate
        acc = __builtin_amdgcn_wmma_f32_16x16x32_bf16(false, ahi, false, bhi, (short)0, acc, false, false);
        acc = __builtin_amdgcn_wmma_f32_16x16x32_bf16(false, ahi, false, blo, (short)0, acc, false, false);
        acc = __builtin_amdgcn_wmma_f32_16x16x32_bf16(false, alo, false, bhi, (short)0, acc, false, false);
    }

    // ======== Phase B: D tile + bias -> LDS ==========================================
    {
        float b1 = b1_mu[h];
#pragma unroll
        for (int r = 0; r < 8; ++r) {
            int m = r + lhalf * 8;                   // C/D layout: M = r + 8*(lane>>4)
            smu[m * MUSTRIDE + h] = acc[r] + b1;     // pre-ReLU mu1
        }
    }
    __syncthreads();

    // ======== Phase C: moment propagation, one wave per batch row ====================
    const int b = mrow + wave;

    // xxt = sum_k x[b,k]^2  (from the LDS-staged tile; pad region is zero)
    float sxx = 0.f;
#pragma unroll
    for (int t = 0; t < 25; ++t) {
        int col = lane + 32 * t;                     // 0..799, pad -> 0
        float xv = sx[wave * XSTRIDE + col];
        sxx += xv * xv;
    }

    float xxt2 = 0.f, sspwg = 0.f, sspbg = 0.f;
    float c0 = 0.f, c1 = 0.f, d0 = 0.f, d1 = 0.f;
    float A00 = 0.f, A01 = 0.f, A11 = 0.f;
    float m20 = 0.f, m21 = 0.f;
#pragma unroll
    for (int t = 0; t < 8; ++t) {
        int i = lane + 32 * t;                       // H index
        float mu  = smu[wave * MUSTRIDE + i];
        float g   = (mu > 0.f) ? 1.f : 0.f;          // relu'(mu)
        float mur = (mu > 0.f) ? mu  : 0.f;          // relu(mu)
        float spw = softplusf(w1_sigma[i]);
        float spb = softplusf(b1_sigma[i]);
        float w20 = w2_mu[2 * i];
        float w21 = w2_mu[2 * i + 1];
        xxt2  += mur * mur;
        sspwg += spw * g;
        sspbg += spb * g;
        c0 += w20 * g;           c1 += w21 * g;
        d0 += spb * g * w20;     d1 += spb * g * w21;
        A00 += w20 * w20 * spw * g;
        A01 += w20 * w21 * spw * g;
        A11 += w21 * w21 * spw * g;
        m20 += mur * w20;        m21 += mur * w21;
    }
    sxx = wred(sxx);     xxt2 = wred(xxt2);
    sspwg = wred(sspwg); sspbg = wred(sspbg);
    c0 = wred(c0); c1 = wred(c1); d0 = wred(d0); d1 = wred(d1);
    A00 = wred(A00); A01 = wred(A01); A11 = wred(A11);
    m20 = wred(m20); m21 = wred(m21);

    if (lane == 0) {
        float mu2_0 = m20 + b2_mu[0];
        float mu2_1 = m21 + b2_mu[1];

        // S_w = xxt * A + c d^T  (A symmetric)
        float Sw00 = sxx * A00 + c0 * d0;
        float Sw01 = sxx * A01 + c0 * d1;
        float Sw10 = sxx * A01 + c1 * d0;
        float Sw11 = sxx * A11 + c1 * d1;

        float trS = sxx * sspwg + sspbg;   // trace(Sigma1 after relu mask)
        float t2  = xxt2 + trS;

        float S00 = Sw00 + t2 * softplusf(w2_sigma[0]) + softplusf(b2_sigma[0]);
        float S01 = Sw01;
        float S10 = Sw10;
        float S11 = Sw11 + t2 * softplusf(w2_sigma[1]) + softplusf(b2_sigma[1]);

        // stable softmax (C = 2)
        float mm = fmaxf(mu2_0, mu2_1);
        float e0 = expf(mu2_0 - mm), e1 = expf(mu2_1 - mm);
        float inv = 1.f / (e0 + e1);
        float p0 = e0 * inv, p1 = e1 * inv;

        // J = diag(p) - p p^T
        float J00 = p0 - p0 * p0, J01 = -p0 * p1;
        float J10 = -p1 * p0,     J11 = p1 - p1 * p1;

        // Sigma_out = J * Sigma2 * J^T
        float T00 = J00 * S00 + J01 * S10;
        float T01 = J00 * S01 + J01 * S11;
        float T10 = J10 * S00 + J11 * S10;
        float T11 = J10 * S01 + J11 * S11;
        float O00 = T00 * J00 + T01 * J01;
        float O01 = T00 * J10 + T01 * J11;
        float O10 = T10 * J00 + T11 * J01;
        float O11 = T10 * J10 + T11 * J11;

        out[2 * b]     = p0;
        out[2 * b + 1] = p1;
        float* so = out + 2 * BATCH + 4 * b;
        so[0] = O00; so[1] = O01; so[2] = O10; so[3] = O11;
    }
}

extern "C" void kernel_launch(void* const* d_in, const int* in_sizes, int n_in,
                              void* d_out, int out_size, void* d_ws, size_t ws_size,
                              hipStream_t stream) {
    (void)in_sizes; (void)n_in; (void)d_ws; (void)ws_size; (void)out_size;
    const float* x        = (const float*)d_in[0];
    const float* w1_mu    = (const float*)d_in[1];
    const float* w1_sigma = (const float*)d_in[2];
    const float* b1_mu    = (const float*)d_in[3];
    const float* b1_sigma = (const float*)d_in[4];
    const float* w2_mu    = (const float*)d_in[5];
    const float* w2_sigma = (const float*)d_in[6];
    const float* b2_mu    = (const float*)d_in[7];
    const float* b2_sigma = (const float*)d_in[8];
    float* out = (float*)d_out;

    exvdp_fused_kernel<<<dim3(BATCH / 16), dim3(512), 0, stream>>>(
        x, w1_mu, w1_sigma, b1_mu, b1_sigma, w2_mu, w2_sigma, b2_mu, b2_sigma, out);
}
